// EGraphSAGELayer_9414568312948
// MI455X (gfx1250) — matrix operand order repaired and checked by reading, not compiled, not run on hardware
//
#include <hip/hip_runtime.h>

// ---------------------------------------------------------------------------
// EGraphSAGE layer for MI455X (gfx1250, wave32, WMMA f32 <= f16 x f16)
//
// inputs : nfeats[50000,128] f32 | efeats[800000,128] f32 | u[800000] i32 |
//          v[800000] i32 | W_apply_w[128,384] f32 | W_apply_b[128] f32 |
//          W_edge_w[128,256] f32 | W_edge_b[128] f32
// outputs: h_nodes_new[50000,128] f32 ++ h_edges_new[800000,128] f32
// ---------------------------------------------------------------------------

typedef __attribute__((ext_vector_type(16))) _Float16 v16h;
typedef __attribute__((ext_vector_type(8)))  _Float16 h8;
typedef __attribute__((ext_vector_type(2)))  _Float16 h2;
typedef __attribute__((ext_vector_type(8)))  float    v8f;

#define N_NODES 50000
#define N_EDGES 800000
#define DIM     128
#define K_NODE  384                 // concat(nfeats, h_neigh[256])
#define K_EDGE  256                 // concat(h[u], h[v])
#define NCH_NODE (K_NODE / 32)      // 12 K-chunks
#define NCH_EDGE (K_EDGE / 32)      // 8 K-chunks
#define EDGE_FRAG_H2 (NCH_EDGE * 8 * 32 * 8)   // 16384 h2 = 64 KB

// ---------------------------------------------------------------------------
// 0) zero the accumulator region in workspace
// ---------------------------------------------------------------------------
__global__ void zero_kernel(float* __restrict__ p, long long n) {
  long long i = (long long)blockIdx.x * blockDim.x + threadIdx.x;
  long long stride = (long long)gridDim.x * blockDim.x;
  for (; i < n; i += stride) p[i] = 0.0f;
}

// ---------------------------------------------------------------------------
// 1) pack W^T into FRAGMENT-MAJOR f16 layout:
//    Wfrag[((c*8 + t)*32 + lane) * 8 + i] = ( W[n][k], W[n][k+1] )
//    with k = c*32 + (lane>>4)*8 + 2i + (i>=4 ? 8 : 0) , n = t*16 + (lane&15)
//    -> each lane's whole 16-half B fragment is one contiguous 32B block
// ---------------------------------------------------------------------------
__global__ void pack_weights_kernel(const float* __restrict__ Wa,  // [128][384]
                                    const float* __restrict__ We,  // [128][256]
                                    h2* __restrict__ Wa_frag,      // 12*8*32*8
                                    h2* __restrict__ We_frag) {    // 8*8*32*8
  const int tid = blockIdx.x * blockDim.x + threadIdx.x;
  const int i    = tid & 7;
  const int lane = (tid >> 3) & 31;
  const int t    = (tid >> 8) & 7;
  const int c    = tid >> 11;
  const int k    = c * 32 + (lane >> 4) * 8 + 2 * i + (i >= 4 ? 8 : 0);
  const int n    = t * 16 + (lane & 15);
  if (tid < NCH_NODE * 8 * 32 * 8) {
    h2 x;
    x.x = (_Float16)Wa[n * K_NODE + k];
    x.y = (_Float16)Wa[n * K_NODE + k + 1];
    Wa_frag[tid] = x;
  }
  if (tid < EDGE_FRAG_H2) {
    h2 x;
    x.x = (_Float16)We[n * K_EDGE + k];
    x.y = (_Float16)We[n * K_EDGE + k + 1];
    We_frag[tid] = x;
  }
}

// ---------------------------------------------------------------------------
// 2) edge message scatter: sums[v] += concat(nfeats[u], efeats[e]); counts[v]++
//    sums (51 MB) is L2-resident -> fp32 atomics stay on-chip
// ---------------------------------------------------------------------------
__global__ void scatter_kernel(const float* __restrict__ nfeats,
                               const float* __restrict__ efeats,
                               const int* __restrict__ u,
                               const int* __restrict__ v,
                               float* __restrict__ sums,     // [N_NODES][256]
                               float* __restrict__ counts) { // [N_NODES]
  const int e = blockIdx.x;
  const int c = threadIdx.x;           // 0..127
  const int src = u[e];
  const int dst = v[e];
  atomicAdd(&sums[(size_t)dst * 256 + c],       nfeats[(size_t)src * DIM + c]);
  atomicAdd(&sums[(size_t)dst * 256 + 128 + c], efeats[(size_t)e   * DIM + c]);
  if (c == 0) atomicAdd(&counts[dst], 1.0f);
}

// ---------------------------------------------------------------------------
// helpers
// ---------------------------------------------------------------------------
// A-fragment K position for 16-bit A 16x32 (ISA 7.12.2):
//   V0..3 : K = 2i + 8*half ,  V4..7 : K = 16 + 2(i-4) + 8*half
__device__ __forceinline__ int a_frag_k(int i, int half_) {
  return (i < 4 ? 2 * i : 16 + 2 * (i - 4)) + (half_ ? 8 : 0);
}

// f16 A-fragment: two contiguous 8-half runs (16B each, 16B-aligned)
__device__ __forceinline__ v16h load_afrag_f16(const _Float16* __restrict__ row,
                                               int kb, int half_) {
  const h8 lo = *(const h8*)(row + kb + 8 * half_);        // V0..3 pairs
  const h8 hi = *(const h8*)(row + kb + 16 + 8 * half_);   // V4..7 pairs
  v16h a;
#pragma unroll
  for (int j = 0; j < 8; ++j) { a[j] = lo[j]; a[8 + j] = hi[j]; }
  return a;
}

// ---------------------------------------------------------------------------
// 3) node update: out = relu([nfeats | sums/count] @ Wa^T + ba)
//    one wave per 16 nodes; all 8 B fragments of a chunk are preloaded into
//    registers so the 16 b128 loads clause together and the 8 WMMAs issue
//    back-to-back (no per-WMMA loadcnt stall). Also emits f16 copy for the
//    edge GEMM.
// ---------------------------------------------------------------------------
__global__ void __launch_bounds__(256)
node_gemm_kernel(const float* __restrict__ nfeats,
                 const float* __restrict__ sums,
                 const float* __restrict__ counts,
                 const h2* __restrict__ Wfrag,       // fragment-major, 12 chunks
                 const float* __restrict__ bias,     // [128]
                 float* __restrict__ out,            // [N_NODES][128] f32
                 _Float16* __restrict__ out16) {     // [N_NODES][128] f16
  const int wave  = blockIdx.x * (blockDim.x >> 5) + (threadIdx.x >> 5);
  const int lane  = threadIdx.x & 31;
  if (wave >= N_NODES / 16) return;                  // wave-uniform
  const int lrow  = lane & 15;
  const int half_ = lane >> 4;
  const int row   = wave * 16 + lrow;

  const float cnt = counts[row];
  const float inv = cnt > 0.0f ? 1.0f / cnt : 0.0f;  // DGL-style mean
  const float* __restrict__ nrow = nfeats + (size_t)row * DIM;
  const float* __restrict__ srow = sums   + (size_t)row * 256;

  v8f zero = {};
  v8f acc[8];
#pragma unroll
  for (int t = 0; t < 8; ++t) acc[t] = zero;

  for (int c = 0; c < NCH_NODE; ++c) {
    const int kc = c * 32;
    // ---- preload all 8 B fragments of this chunk (16 clause-able b128) ----
    v16h bfrag[8];
#pragma unroll
    for (int t = 0; t < 8; ++t)
      bfrag[t] = *(const v16h*)&Wfrag[(size_t)((c * 8 + t) * 32 + lane) * 8];

    // ---- A fragment (fp32 sources, mean folded in, cvt to f16) ----
    v16h a;
#pragma unroll
    for (int i = 0; i < 8; ++i) {
      const int k = kc + a_frag_k(i, half_);
      float f0, f1;
      if (k < DIM) { f0 = nrow[k];             f1 = nrow[k + 1]; }
      else         { f0 = srow[k - DIM] * inv; f1 = srow[k - DIM + 1] * inv; }
      a[2 * i]     = (_Float16)f0;
      a[2 * i + 1] = (_Float16)f1;
    }

    // ---- 8 back-to-back WMMAs ----
#pragma unroll
    for (int t = 0; t < 8; ++t) {
      acc[t] = __builtin_amdgcn_wmma_f32_16x16x32_f16(
          false, a, false, bfrag[t], (short)0, acc[t], false, false);
    }
  }

  // ---- C/D layout: element (M = r + 8*half, N = lane%16) in VGPR r ----
#pragma unroll
  for (int t = 0; t < 8; ++t) {
    const int n = t * 16 + lrow;
    const float bn = bias[n];
#pragma unroll
    for (int r = 0; r < 8; ++r) {
      const int m = wave * 16 + r + 8 * half_;
      float val = acc[t][r] + bn;
      val = val > 0.0f ? val : 0.0f;
      out[(size_t)m * DIM + n]   = val;
      out16[(size_t)m * DIM + n] = (_Float16)val;
    }
  }
}

// ---------------------------------------------------------------------------
// 4) edge update: out = relu([h[u] | h[v]] @ We^T + be)
//    one wave per 32 edges (2 row-tiles share every B fragment);
//    B fragments staged once per block in LDS (64 KB) -> ds_load_b128 in the
//    hot loop instead of latency-bound global loads; A operands are f16.
// ---------------------------------------------------------------------------
__global__ void __launch_bounds__(256)
edge_gemm_kernel(const _Float16* __restrict__ h16,   // [N_NODES][128] f16
                 const int* __restrict__ u,
                 const int* __restrict__ v,
                 const h2* __restrict__ Wfrag,       // fragment-major, 8 chunks
                 const float* __restrict__ bias,     // [128]
                 float* __restrict__ out) {          // [N_EDGES][128]
  __shared__ __align__(32) h2 Bsh[EDGE_FRAG_H2];     // 64 KB, shared by 8 waves

  // ---- cooperative stage: 4096 x 16B ----
  {
    const uint4* __restrict__ src = (const uint4*)Wfrag;
    uint4* dst = (uint4*)Bsh;
    for (int idx = threadIdx.x; idx < EDGE_FRAG_H2 / 4; idx += blockDim.x)
      dst[idx] = src[idx];
  }
  __syncthreads();

  const int wave  = blockIdx.x * (blockDim.x >> 5) + (threadIdx.x >> 5);
  const int lane  = threadIdx.x & 31;
  if (wave >= N_EDGES / 32) return;                  // grid is exact; safety only
  const int lrow  = lane & 15;
  const int half_ = lane >> 4;
  const int base  = wave * 32;

  const int e0 = base + lrow;        // row-tile 0
  const int e1 = base + 16 + lrow;   // row-tile 1
  const _Float16* __restrict__ ur0 = h16 + (size_t)u[e0] * DIM;
  const _Float16* __restrict__ vr0 = h16 + (size_t)v[e0] * DIM;
  const _Float16* __restrict__ ur1 = h16 + (size_t)u[e1] * DIM;
  const _Float16* __restrict__ vr1 = h16 + (size_t)v[e1] * DIM;

  v8f zero = {};
  v8f acc0[8], acc1[8];
#pragma unroll
  for (int t = 0; t < 8; ++t) { acc0[t] = zero; acc1[t] = zero; }

  for (int c = 0; c < NCH_EDGE; ++c) {
    const int kc = c * 32;
    const bool useU = kc < DIM;                 // concat boundary is chunk-aligned
    const int  kb   = kc & (DIM - 1);
    const v16h a0 = load_afrag_f16(useU ? ur0 : vr0, kb, half_);
    const v16h a1 = load_afrag_f16(useU ? ur1 : vr1, kb, half_);
#pragma unroll
    for (int t = 0; t < 8; ++t) {
      const v16h b = *(const v16h*)&Bsh[(size_t)((c * 8 + t) * 32 + lane) * 8];
      acc0[t] = __builtin_amdgcn_wmma_f32_16x16x32_f16(
          false, a0, false, b, (short)0, acc0[t], false, false);
      acc1[t] = __builtin_amdgcn_wmma_f32_16x16x32_f16(
          false, a1, false, b, (short)0, acc1[t], false, false);
    }
  }

#pragma unroll
  for (int t = 0; t < 8; ++t) {
    const int n = t * 16 + lrow;
    const float bn = bias[n];
#pragma unroll
    for (int r = 0; r < 8; ++r) {
      const int m = r + 8 * half_;
      float v0 = acc0[t][r] + bn;
      float v1 = acc1[t][r] + bn;
      out[(size_t)(base + m) * DIM + n]      = v0 > 0.0f ? v0 : 0.0f;
      out[(size_t)(base + 16 + m) * DIM + n] = v1 > 0.0f ? v1 : 0.0f;
    }
  }
}

// ---------------------------------------------------------------------------
// launch
// ---------------------------------------------------------------------------
extern "C" void kernel_launch(void* const* d_in, const int* in_sizes, int n_in,
                              void* d_out, int out_size, void* d_ws, size_t ws_size,
                              hipStream_t stream) {
  (void)in_sizes; (void)n_in; (void)out_size; (void)ws_size;

  const float* nfeats = (const float*)d_in[0];
  const float* efeats = (const float*)d_in[1];
  const int*   u      = (const int*)d_in[2];
  const int*   v      = (const int*)d_in[3];
  const float* Wa     = (const float*)d_in[4];
  const float* ba     = (const float*)d_in[5];
  const float* We     = (const float*)d_in[6];
  const float* be     = (const float*)d_in[7];

  float* out_nodes = (float*)d_out;                          // [N_NODES*128]
  float* out_edges = out_nodes + (size_t)N_NODES * DIM;      // [N_EDGES*128]

  // workspace: sums | counts | Wa_frag | We_frag | h16   (~64.5 MB total)
  char* ws = (char*)d_ws;
  size_t off = 0;
  float* sums = (float*)(ws + off);   off += (size_t)N_NODES * 256 * sizeof(float);
  float* counts = (float*)(ws + off); off += (size_t)N_NODES * sizeof(float);
  off = (off + 255) & ~(size_t)255;
  h2* Wa_frag = (h2*)(ws + off);      off += (size_t)NCH_NODE * 8 * 32 * 8 * sizeof(h2);
  off = (off + 255) & ~(size_t)255;
  h2* We_frag = (h2*)(ws + off);      off += (size_t)EDGE_FRAG_H2 * sizeof(h2);
  off = (off + 255) & ~(size_t)255;
  _Float16* h16 = (_Float16*)(ws + off);

  // 0) zero accumulators (sums ++ counts are contiguous)
  zero_kernel<<<2048, 256, 0, stream>>>(sums, (long long)N_NODES * 256 + N_NODES);

  // 1) pack weights to fragment-major f16
  pack_weights_kernel<<<(NCH_NODE * 8 * 32 * 8 + 255) / 256, 256, 0, stream>>>(
      Wa, We, Wa_frag, We_frag);

  // 2) edge message scatter (L2-resident atomics)
  scatter_kernel<<<N_EDGES, DIM, 0, stream>>>(nfeats, efeats, u, v, sums, counts);

  // 3) node GEMM+ReLU: 3125 row-tiles, 8 waves/block
  node_gemm_kernel<<<(N_NODES / 16 + 7) / 8, 256, 0, stream>>>(
      nfeats, sums, counts, Wa_frag, ba, out_nodes, h16);

  // 4) edge GEMM+ReLU: 25000 waves (32 rows each), 8 waves/block, LDS-staged B
  edge_gemm_kernel<<<(N_EDGES / 32) / 8, 256, 0, stream>>>(
      h16, u, v, We_frag, be, out_edges);
}
